// DynamicGNN_30751965839948
// MI455X (gfx1250) — matrix operand order repaired and checked by reading, not compiled, not run on hardware
//
#include <hip/hip_runtime.h>
#include <hip/hip_bf16.h>

typedef __attribute__((ext_vector_type(16))) _Float16 v16h;
typedef __attribute__((ext_vector_type(8)))  _Float16 v8h;
typedef __attribute__((ext_vector_type(8)))  float    v8f;

#define NNODES 39
#define KNN    10
#define HID    256
#define NGRAPH 512
#define MROWS  (NGRAPH * NNODES)   // 19968, divisible by 64
#define NCOLS  512                 // [p | q]
#define EPSBN  1e-5f

// ---------------------------------------------------------------------------
// kNN: one workgroup per graph. Distances d_ij = |xi|^2+|xj|^2-2 xi.xj in LDS,
// selection-style top-K (strict < keeps earliest index == jax top_k ties).
// For layers 2/3 the previous layer's BN+ReLU is applied on load.
// ---------------------------------------------------------------------------
__global__ void knn_kernel(const float* __restrict__ X, int F, int applyBN,
                           const float* __restrict__ stats, int* __restrict__ idx) {
  __shared__ float xl[NNODES * 256];
  __shared__ float sq[NNODES];
  __shared__ float dist[NNODES * NNODES];
  __shared__ float sc[256], sh[256];
  const int g = blockIdx.x, tid = threadIdx.x;

  if (applyBN && tid < 256) { sc[tid] = stats[tid]; sh[tid] = stats[256 + tid]; }
  __syncthreads();

  const float* xg = X + (size_t)g * NNODES * F;
  for (int e = tid; e < NNODES * F; e += 256) {
    float v = xg[e];
    if (applyBN) { int f = e & 255; v = fmaxf(v * sc[f] + sh[f], 0.f); } // F==256 here
    xl[e] = v;
  }
  __syncthreads();

  if (tid < NNODES) {
    float s = 0.f;
    for (int f = 0; f < F; ++f) { float v = xl[tid * F + f]; s += v * v; }
    sq[tid] = s;
  }
  __syncthreads();

  for (int p = tid; p < NNODES * NNODES; p += 256) {
    int i = p / NNODES, j = p % NNODES;
    const float* xi = xl + i * F;
    const float* xj = xl + j * F;
    float dot = 0.f;
    for (int f = 0; f < F; ++f) dot += xi[f] * xj[f];
    dist[p] = sq[i] + sq[j] - 2.f * dot;
  }
  __syncthreads();

  if (tid < NNODES) {
    float* drow = dist + tid * NNODES;
    for (int k = 0; k < KNN; ++k) {
      float best = 3.4e38f; int bj = 0;
      for (int j = 0; j < NNODES; ++j) { float d = drow[j]; if (d < best) { best = d; bj = j; } }
      drow[bj] = 3.4e38f;
      idx[((size_t)g * NNODES + tid) * KNN + k] = g * NNODES + bj;
    }
  }
}

// ---------------------------------------------------------------------------
// prepA: activation (f32, optional BN+ReLU) -> f16, zero-padded to Kpad.
// ---------------------------------------------------------------------------
__global__ void prepA_kernel(const float* __restrict__ X, int F, int kshift,
                             int applyBN, const float* __restrict__ stats,
                             _Float16* __restrict__ Ah) {
  int t = blockIdx.x * 256 + threadIdx.x;         // over MROWS << kshift
  int mask = (1 << kshift) - 1;
  int k = t & mask;
  int row = t >> kshift;
  float v = 0.f;
  if (k < F) {
    v = X[(size_t)row * F + k];
    if (applyBN) v = fmaxf(v * stats[k] + stats[256 + k], 0.f);
  }
  Ah[t] = (_Float16)v;
}

// ---------------------------------------------------------------------------
// prepW: build combined weight [D | W_bot] (D = W_top - W_bot), f16, directly
// in the per-lane WMMA B-fragment layout:
//   Bh[((ct*KC + kc)*32 + lane)*16 + v], lane: n = ct*16 + lane%16, hl=lane/16,
//   K index = kc*32 + hl*16 + v   (B 32x16 f16: 16 contiguous K per lane).
// ---------------------------------------------------------------------------
__global__ void prepW_kernel(const float* __restrict__ W, int fin, int Kpad,
                             _Float16* __restrict__ Bh) {
  int t = blockIdx.x * 256 + threadIdx.x;         // over Kpad*512
  int v = t & 15;
  int l = (t >> 4) & 31;
  int chunk = t >> 9;                             // ct*KC + kc
  int KC = Kpad >> 5;
  int ct = chunk / KC;
  int kc = chunk - ct * KC;
  int hl = l >> 4;
  int n = ct * 16 + (l & 15);
  int kk = kc * 32 + hl * 16 + v;
  float val = 0.f;
  if (kk < fin) {
    if (n < 256) val = W[kk * 256 + n] - W[(fin + kk) * 256 + n];   // W_top - W_bot
    else         val = W[(fin + kk) * 256 + (n - 256)];             // W_bot
  }
  Bh[t] = (_Float16)val;
}

// ---------------------------------------------------------------------------
// WMMA GEMM: pq[M x 512] = Ah[M x Kpad] @ Bh[Kpad x 512], f16 in / f32 acc.
// Register-tiled: each wave computes a 64(M) x 32(N) block = 4x2 WMMA tiles,
// so each loaded fragment feeds multiple WMMAs (1.5 b128 loads per WMMA
// instead of 4) -> ~2.7x less L2 fragment traffic.
// A fragment (16x32 f16): lane holds row m = l%16, 16 contiguous K at
// K = kc*32 + hl*16 + [0..16) (matches B fragment layout).
// ---------------------------------------------------------------------------
#define GMT 4   // M tiles per wave
#define GNT 2   // N tiles per wave
__global__ void gemm_kernel(const _Float16* __restrict__ Ah,
                            const _Float16* __restrict__ Bh,
                            float* __restrict__ pq, int Kpad) {
  const int KC = Kpad >> 5;
  int wave = (blockIdx.x * 256 + threadIdx.x) >> 5; // 4992 waves total
  int lane = threadIdx.x & 31;
  int ms = wave >> 4;              // M super-tile (0..311), rows ms*64..
  int cs = wave & 15;              // N super-tile (0..15), cols cs*32..
  int hl = lane >> 4;
  int ml = lane & 15;

  const _Float16* arow0 = Ah + (size_t)(ms * 64 + ml) * Kpad;     // mi stride 16*Kpad
  const _Float16* bp0   = Bh + ((size_t)(cs * GNT) * KC * 32 + lane) * 16; // ni stride KC*32*16

  v8f acc[GMT][GNT];
  #pragma unroll
  for (int mi = 0; mi < GMT; ++mi)
    #pragma unroll
    for (int ni = 0; ni < GNT; ++ni)
      acc[mi][ni] = (v8f){0.f, 0.f, 0.f, 0.f, 0.f, 0.f, 0.f, 0.f};

  for (int kc = 0; kc < KC; ++kc) {
    v16h a[GMT];
    #pragma unroll
    for (int mi = 0; mi < GMT; ++mi) {
      const _Float16* ar = arow0 + (size_t)mi * 16 * Kpad + kc * 32;
      a[mi] = *(const v16h*)(ar + hl * 16);
    }
    v16h b[GNT];
    #pragma unroll
    for (int ni = 0; ni < GNT; ++ni)
      b[ni] = *(const v16h*)(bp0 + ((size_t)ni * KC + kc) * 32 * 16);

    #pragma unroll
    for (int mi = 0; mi < GMT; ++mi)
      #pragma unroll
      for (int ni = 0; ni < GNT; ++ni)
        acc[mi][ni] = __builtin_amdgcn_wmma_f32_16x16x32_f16(
            false, a[mi], false, b[ni], (short)0, acc[mi][ni], false, false);
  }

  #pragma unroll
  for (int mi = 0; mi < GMT; ++mi) {
    int rowbase = ms * 64 + mi * 16 + hl * 8;
    #pragma unroll
    for (int ni = 0; ni < GNT; ++ni) {
      int n = cs * 32 + ni * 16 + ml;
      #pragma unroll
      for (int r = 0; r < 8; ++r)
        pq[(size_t)(rowbase + r) * NCOLS + n] = acc[mi][ni][r];
    }
  }
}

// ---------------------------------------------------------------------------
// Aggregation: h[i,c] = relu(max_k (p_i[c] + q_{idx[i,k]}[c] + b[c]))
// (relu(max) == max(relu) since relu is monotone and K>=1).
// ---------------------------------------------------------------------------
__global__ void agg_kernel(const float* __restrict__ pq, const int* __restrict__ idx,
                           const float* __restrict__ bias, float* __restrict__ h) {
  int i = blockIdx.x, c = threadIdx.x;
  float base = pq[(size_t)i * NCOLS + c] + bias[c];
  const int* nb = idx + (size_t)i * KNN;
  float best = -3.4e38f;
  #pragma unroll
  for (int k = 0; k < KNN; ++k) {
    int j = nb[k];
    best = fmaxf(best, base + pq[(size_t)j * NCOLS + 256 + c]);
  }
  h[(size_t)i * HID + c] = fmaxf(best, 0.f);
}

// ---------------------------------------------------------------------------
// Deterministic BN statistics: stage 1, 256 blocks x 78 rows each.
// ---------------------------------------------------------------------------
__global__ void reduce_kernel(const float* __restrict__ h, float* __restrict__ partial) {
  int b = blockIdx.x, c = threadIdx.x;
  const float* base = h + (size_t)b * 78 * HID;
  float s = 0.f, s2 = 0.f;
  for (int r = 0; r < 78; ++r) { float v = base[r * HID + c]; s += v; s2 += v * v; }
  partial[(size_t)b * 512 + c] = s;
  partial[(size_t)b * 512 + 256 + c] = s2;
}

__global__ void bnparams_kernel(const float* __restrict__ partial,
                                const float* __restrict__ gamma,
                                const float* __restrict__ beta,
                                float* __restrict__ stats) {
  int c = threadIdx.x;
  float s = 0.f, s2 = 0.f;
  for (int b = 0; b < 256; ++b) { s += partial[b * 512 + c]; s2 += partial[b * 512 + 256 + c]; }
  const float invM = 1.f / (float)MROWS;
  float mean = s * invM;
  float var = s2 * invM - mean * mean;
  float scale = gamma[c] * rsqrtf(var + EPSBN);
  stats[c] = scale;
  stats[256 + c] = beta[c] - mean * scale;
}

// ---------------------------------------------------------------------------
// Head: BN3 (no relu) + mean pool over nodes + 256->2 linear + softmax.
// ---------------------------------------------------------------------------
__global__ void head_kernel(const float* __restrict__ h, const float* __restrict__ stats,
                            const float* __restrict__ Wl, const float* __restrict__ bl,
                            float* __restrict__ out) {
  __shared__ float r0[256], r1[256];
  int g = blockIdx.x, c = threadIdx.x;
  float scale = stats[c], shift = stats[256 + c];
  const float* base = h + (size_t)g * NNODES * HID;
  float s = 0.f;
  for (int n = 0; n < NNODES; ++n) s += base[n * HID + c];
  float pooled = s * scale * (1.f / (float)NNODES) + shift;
  r0[c] = pooled * Wl[c * 2 + 0];
  r1[c] = pooled * Wl[c * 2 + 1];
  __syncthreads();
  for (int st = 128; st > 0; st >>= 1) {
    if (c < st) { r0[c] += r0[c + st]; r1[c] += r1[c + st]; }
    __syncthreads();
  }
  if (c == 0) {
    float l0 = r0[0] + bl[0], l1 = r1[0] + bl[1];
    float m = fmaxf(l0, l1);
    float e0 = expf(l0 - m), e1 = expf(l1 - m);
    float inv = 1.f / (e0 + e1);
    out[g * 2 + 0] = e0 * inv;
    out[g * 2 + 1] = e1 * inv;
  }
}

extern "C" void kernel_launch(void* const* d_in, const int* in_sizes, int n_in,
                              void* d_out, int out_size, void* d_ws, size_t ws_size,
                              hipStream_t stream) {
  (void)in_sizes; (void)n_in; (void)out_size; (void)ws_size;
  const float* x   = (const float*)d_in[0];
  const float* W1  = (const float*)d_in[2];
  const float* b1  = (const float*)d_in[3];
  const float* W2  = (const float*)d_in[4];
  const float* b2  = (const float*)d_in[5];
  const float* W3  = (const float*)d_in[6];
  const float* b3  = (const float*)d_in[7];
  const float* g1  = (const float*)d_in[8];
  const float* be1 = (const float*)d_in[9];
  const float* g2  = (const float*)d_in[10];
  const float* be2 = (const float*)d_in[11];
  const float* g3  = (const float*)d_in[12];
  const float* be3 = (const float*)d_in[13];
  const float* Wl  = (const float*)d_in[14];
  const float* bl  = (const float*)d_in[15];
  float* out = (float*)d_out;

  // Workspace layout (all 256B aligned). Total ~73.2 MB — L2 resident on MI455X.
  char* ws = (char*)d_ws;
  float*    stats   = (float*)(ws + 0);          // scale[256], shift[256]
  float*    partial = (float*)(ws + 4096);       // 256 blocks * 512 f32 = 512 KB
  int*      idx     = (int*)(ws + 528384);       // 199,680 int
  _Float16* Ah      = (_Float16*)(ws + 1327104); // M*256 f16 = 10.2 MB
  _Float16* Bh      = (_Float16*)(ws + 11550720);// 256*512 f16 = 256 KB
  float*    pq      = (float*)(ws + 11812864);   // M*512 f32 = 40.9 MB
  float*    h       = (float*)(ws + 52707328);   // M*256 f32 = 20.4 MB

  // 4992 waves (312 M-super x 16 N-super), 8 waves/block -> 624 blocks
  const int GEMM_BLOCKS = (MROWS / 64) * (NCOLS / 32) / 8;

  // ---- Layer 1 (F=39, Kpad=64, no input BN) ----
  knn_kernel  <<<NGRAPH, 256, 0, stream>>>(x, 39, 0, stats, idx);
  prepA_kernel<<<(MROWS * 64) / 256, 256, 0, stream>>>(x, 39, 6, 0, stats, Ah);
  prepW_kernel<<<(64 * 512) / 256, 256, 0, stream>>>(W1, 39, 64, Bh);
  gemm_kernel <<<GEMM_BLOCKS, 256, 0, stream>>>(Ah, Bh, pq, 64);
  agg_kernel  <<<MROWS, 256, 0, stream>>>(pq, idx, b1, h);
  reduce_kernel  <<<256, 256, 0, stream>>>(h, partial);
  bnparams_kernel<<<1, 256, 0, stream>>>(partial, g1, be1, stats);

  // ---- Layer 2 (F=256, Kpad=256, fused BN1+ReLU on load) ----
  knn_kernel  <<<NGRAPH, 256, 0, stream>>>(h, 256, 1, stats, idx);
  prepA_kernel<<<(MROWS * 256) / 256, 256, 0, stream>>>(h, 256, 8, 1, stats, Ah);
  prepW_kernel<<<(256 * 512) / 256, 256, 0, stream>>>(W2, 256, 256, Bh);
  gemm_kernel <<<GEMM_BLOCKS, 256, 0, stream>>>(Ah, Bh, pq, 256);
  agg_kernel  <<<MROWS, 256, 0, stream>>>(pq, idx, b2, h);
  reduce_kernel  <<<256, 256, 0, stream>>>(h, partial);
  bnparams_kernel<<<1, 256, 0, stream>>>(partial, g2, be2, stats);

  // ---- Layer 3 ----
  knn_kernel  <<<NGRAPH, 256, 0, stream>>>(h, 256, 1, stats, idx);
  prepA_kernel<<<(MROWS * 256) / 256, 256, 0, stream>>>(h, 256, 8, 1, stats, Ah);
  prepW_kernel<<<(256 * 512) / 256, 256, 0, stream>>>(W3, 256, 256, Bh);
  gemm_kernel <<<GEMM_BLOCKS, 256, 0, stream>>>(Ah, Bh, pq, 256);
  agg_kernel  <<<MROWS, 256, 0, stream>>>(pq, idx, b3, h);
  reduce_kernel  <<<256, 256, 0, stream>>>(h, partial);
  bnparams_kernel<<<1, 256, 0, stream>>>(partial, g3, be3, stats);

  // ---- Head: BN3 + mean pool + linear + softmax ----
  head_kernel<<<NGRAPH, 256, 0, stream>>>(h, stats, Wl, bl, out);
}
